// MakeupFluxAttnProcessor_59931973648604
// MI455X (gfx1250) — matrix operand order repaired and checked by reading, not compiled, not run on hardware
//
#include <hip/hip_runtime.h>
#include <hip/hip_bf16.h>
#include <cstddef>
#include <cstdint>

#define HEADS 24
#define HDIM  128
#define INNER 3072
#define RANK  1024
#define S_IMG 2048
#define S_TXT 512
#define S_REF 2048
#define SQ_TOT (S_TXT + S_IMG)          /* 2560 */
#define SK_TOT (S_TXT + S_IMG + S_REF)  /* 4608 */

typedef __attribute__((ext_vector_type(16))) __bf16 v16bf;
typedef __attribute__((ext_vector_type(8)))  float  v8f;

union Frag { v16bf v; unsigned int u[8]; };

__device__ __forceinline__ unsigned short f32_bf16(float f) {
  unsigned int u = __float_as_uint(f);
  u += 0x7FFFu + ((u >> 16) & 1u);          // round-to-nearest-even
  return (unsigned short)(u >> 16);
}

// CDNA5 async global->LDS 16B copy (ASYNCcnt-tracked, bypasses VGPRs).
// vdst = per-lane LDS byte offset, vaddr = per-lane 64-bit global address.
__device__ __forceinline__ void async_copy_b128(unsigned lds_off, const void* gaddr) {
  asm volatile("global_load_async_to_lds_b128 %0, %1, off"
               :: "v"(lds_off), "v"(gaddr) : "memory");
}
__device__ __forceinline__ void async_wait0() {
  asm volatile("s_wait_asynccnt 0" ::: "memory");
}
__device__ __forceinline__ unsigned lds_off_of(const void* p) {
  return (unsigned)(uintptr_t)p;   // generic addr of __shared__: low 32 bits = LDS offset
}

// ---------------------------------------------------------------- convert
__global__ void k_cvt_bf16(const float* __restrict__ src,
                           unsigned short* __restrict__ dst, long long n) {
  long long i = (long long)blockIdx.x * blockDim.x + threadIdx.x;
  if (i < n) dst[i] = f32_bf16(src[i]);
}

// ---------------------------------------------------------------- GEMM
// C[M,N] f32 = A[M,K]bf16 @ B[K,N]bf16 (+bias).  128x128x32 tiles, 8 waves.
__global__ __launch_bounds__(256) void k_gemm(
    const unsigned short* __restrict__ A,
    const unsigned short* __restrict__ B,
    const float* __restrict__ bias,
    float* __restrict__ C, int M, int N, int K) {
  __shared__ alignas(16) unsigned short sa[2][128 * 32]; // [m][k]
  __shared__ alignas(16) unsigned short sb[2][128 * 32]; // [n][k] (transposed)
  const int bm = blockIdx.y * 128, bn = blockIdx.x * 128;
  const int tid = threadIdx.x;
  const int lane = tid & 31, wave = tid >> 5;
  const int wr = wave & 3, wc = wave >> 2;   // wave tile: rows wr*32, cols wc*64
  const int r = lane & 15, h = lane >> 4;

  v8f acc[2][4];
  for (int s = 0; s < 2; ++s)
    for (int t = 0; t < 4; ++t)
      acc[s][t] = (v8f){0.f,0.f,0.f,0.f,0.f,0.f,0.f,0.f};

  const int nk = K >> 5;

  auto stage = [&](int k0, int buf) {
#pragma unroll
    for (int ci = 0; ci < 2; ++ci) {
      int c = tid + ci * 256;
      // A: 128 rows x 32 k, straight 16B copies -> async global->LDS path
      int m = c >> 2, ko = (c & 3) << 3;
      async_copy_b128(lds_off_of(&sa[buf][m * 32 + ko]),
                      A + (size_t)(bm + m) * K + k0 + ko);
      // B: coalesced global read along n, transposed scalar stores to [n][k]
      int kk = c >> 4, no = (c & 15) << 3;
      uint4 bv4 = *reinterpret_cast<const uint4*>(B + (size_t)(k0 + kk) * N + bn + no);
      const unsigned short* bs = reinterpret_cast<const unsigned short*>(&bv4);
#pragma unroll
      for (int j = 0; j < 8; ++j) sb[buf][(no + j) * 32 + kk] = bs[j];
    }
  };

  stage(0, 0);
  async_wait0();
  __syncthreads();
  for (int kb = 0; kb < nk; ++kb) {
    int buf = kb & 1;
    if (kb + 2 < nk) {
      __builtin_prefetch(A + (size_t)(bm + (tid & 127)) * K + ((kb + 2) << 5), 0, 3);
      __builtin_prefetch(B + (size_t)(((kb + 2) << 5) + (tid & 31)) * N + bn, 0, 3);
    }
    if (kb + 1 < nk) stage((kb + 1) << 5, buf ^ 1);

    Frag af[2], bf[4];
#pragma unroll
    for (int s = 0; s < 2; ++s) {
      int row = wr * 32 + s * 16 + r;
#pragma unroll
      for (int i = 0; i < 8; ++i) {
        int kk = 2 * i + ((i >= 4) ? 8 : 0) + 8 * h;   // ISA A-frag layout
        af[s].u[i] = *reinterpret_cast<const unsigned int*>(&sa[buf][row * 32 + kk]);
      }
    }
#pragma unroll
    for (int t = 0; t < 4; ++t) {
      int col = wc * 64 + t * 16 + r;
#pragma unroll
      for (int i = 0; i < 8; ++i) {
        int kk = 2 * i + ((i >= 4) ? 8 : 0) + 8 * h;   // ISA B-frag layout
        bf[t].u[i] = *reinterpret_cast<const unsigned int*>(&sb[buf][col * 32 + kk]);
      }
    }
#pragma unroll
    for (int s = 0; s < 2; ++s)
#pragma unroll
      for (int t = 0; t < 4; ++t)
        acc[s][t] = __builtin_amdgcn_wmma_f32_16x16x32_bf16(
            false, af[s].v, false, bf[t].v, (short)0, acc[s][t], false, false);

    async_wait0();      // staged A tile (next buf) resident before handoff
    __syncthreads();
  }
#pragma unroll
  for (int s = 0; s < 2; ++s)
#pragma unroll
    for (int t = 0; t < 4; ++t) {
      int col = bn + wc * 64 + t * 16 + r;
      float bvl = bias ? bias[col] : 0.f;
#pragma unroll
      for (int j = 0; j < 8; ++j) {
        int row = bm + wr * 32 + s * 16 + j + 8 * h;   // ISA C layout
        C[(size_t)row * N + col] = acc[s][t][j] + bvl;
      }
    }
}

// ---------------------------------------------------------------- RMS/RoPE pack
__global__ __launch_bounds__(256) void k_pack(
    const float* __restrict__ X, const float* __restrict__ w,
    const float* __restrict__ cs, const float* __restrict__ sn,
    int rope_off, float eps, float scale, int S, int do_rms,
    unsigned short* __restrict__ dst, int dst_seq, int dst_off) {
  int gw = (blockIdx.x * 256 + threadIdx.x) >> 5;   // one wave per (token, head)
  int lane = threadIdx.x & 31;
  if (gw >= S * HEADS) return;
  int s = gw / HEADS, hd = gw % HEADS;
  const float* x = X + (size_t)s * INNER + hd * HDIM;
  int d0 = lane * 4;
  float v0 = x[d0], v1 = x[d0 + 1], v2 = x[d0 + 2], v3 = x[d0 + 3];
  if (do_rms) {
    float ss = v0 * v0 + v1 * v1 + v2 * v2 + v3 * v3;
    ss += __shfl_xor(ss, 1, 32);  ss += __shfl_xor(ss, 2, 32);
    ss += __shfl_xor(ss, 4, 32);  ss += __shfl_xor(ss, 8, 32);
    ss += __shfl_xor(ss, 16, 32);
    float rsv = rsqrtf(ss * (1.0f / HDIM) + eps);
    v0 *= rsv; v1 *= rsv; v2 *= rsv; v3 *= rsv;
    if (w) { v0 *= w[d0]; v1 *= w[d0 + 1]; v2 *= w[d0 + 2]; v3 *= w[d0 + 3]; }
  }
  if (cs) {
    const float* c = cs + (size_t)(rope_off + s) * HDIM + d0;
    const float* si = sn + (size_t)(rope_off + s) * HDIM + d0;
    float o0 = v0 * c[0] - v1 * si[0];
    float o1 = v1 * c[1] + v0 * si[1];
    float o2 = v2 * c[2] - v3 * si[2];
    float o3 = v3 * c[3] + v2 * si[3];
    v0 = o0; v1 = o1; v2 = o2; v3 = o3;
  }
  v0 *= scale; v1 *= scale; v2 *= scale; v3 *= scale;
  unsigned short* dp = dst + ((size_t)hd * dst_seq + dst_off + s) * HDIM + d0;
  dp[0] = f32_bf16(v0); dp[1] = f32_bf16(v1);
  dp[2] = f32_bf16(v2); dp[3] = f32_bf16(v3);
}

// ---------------------------------------------------------------- flash attention
// 4 waves/block, 16 q rows/wave, K/V tiles of 32 double-buffered in LDS via
// CDNA5 async global->LDS copies.
__global__ __launch_bounds__(128) void k_attn(
    const unsigned short* __restrict__ Qb,
    const unsigned short* __restrict__ Kb,
    const unsigned short* __restrict__ Vb,
    float* __restrict__ Out) {
  __shared__ alignas(16) unsigned short sk[2][32 * 128];
  __shared__ alignas(16) unsigned short sv[2][32 * 128];
  __shared__ alignas(16) unsigned short sp[4][16 * 32];
  const int head = blockIdx.y;
  const int tid = threadIdx.x;
  const int lane = tid & 31, wave = tid >> 5;
  const int r = lane & 15, h = lane >> 4;
  const int q0 = blockIdx.x * 64 + wave * 16;

  const unsigned short* Qh = Qb + ((size_t)head * SQ_TOT + q0) * HDIM;
  const unsigned short* Kh = Kb + (size_t)head * SK_TOT * HDIM;
  const unsigned short* Vh = Vb + (size_t)head * SK_TOT * HDIM;

  Frag qf[4];
#pragma unroll
  for (int c = 0; c < 4; ++c)
#pragma unroll
    for (int i = 0; i < 8; ++i) {
      int kk = c * 32 + 2 * i + ((i >= 4) ? 8 : 0) + 8 * h;
      qf[c].u[i] = *reinterpret_cast<const unsigned int*>(Qh + (size_t)r * HDIM + kk);
    }

  v8f o[8];
  float mrow[8], lrow[8];
#pragma unroll
  for (int t = 0; t < 8; ++t) o[t] = (v8f){0.f,0.f,0.f,0.f,0.f,0.f,0.f,0.f};
#pragma unroll
  for (int j = 0; j < 8; ++j) { mrow[j] = -1e30f; lrow[j] = 0.f; }

  auto stage = [&](int kb, int buf) {
    const unsigned short* ks = Kh + (size_t)kb * 32 * HDIM;
    const unsigned short* vs = Vh + (size_t)kb * 32 * HDIM;
#pragma unroll
    for (int ci = 0; ci < 4; ++ci) {
      int c = tid + ci * 128;
      int row = c >> 4, off = (c & 15) << 3;
      async_copy_b128(lds_off_of(&sk[buf][row * 128 + off]),
                      ks + (size_t)row * HDIM + off);
      async_copy_b128(lds_off_of(&sv[buf][row * 128 + off]),
                      vs + (size_t)row * HDIM + off);
    }
  };

  const int nkb = SK_TOT / 32;   // 144, exact
  stage(0, 0);
  async_wait0();
  __syncthreads();
  for (int kb = 0; kb < nkb; ++kb) {
    int buf = kb & 1;
    if (kb + 2 < nkb) {
      __builtin_prefetch(Kh + ((size_t)(kb + 2) * 32) * HDIM + tid * 32, 0, 3);
      __builtin_prefetch(Vh + ((size_t)(kb + 2) * 32) * HDIM + tid * 32, 0, 3);
    }
    if (kb + 1 < nkb) stage(kb + 1, buf ^ 1);

    // S = Q @ K^T for two 16-wide key tiles (scale pre-folded into Q)
    v8f s0 = (v8f){0.f,0.f,0.f,0.f,0.f,0.f,0.f,0.f};
    v8f s1 = (v8f){0.f,0.f,0.f,0.f,0.f,0.f,0.f,0.f};
#pragma unroll
    for (int c = 0; c < 4; ++c) {
      Frag kf0, kf1;
#pragma unroll
      for (int i = 0; i < 8; ++i) {
        int d = c * 32 + 2 * i + ((i >= 4) ? 8 : 0) + 8 * h;
        kf0.u[i] = *reinterpret_cast<const unsigned int*>(&sk[buf][r * 128 + d]);
        kf1.u[i] = *reinterpret_cast<const unsigned int*>(&sk[buf][(16 + r) * 128 + d]);
      }
      s0 = __builtin_amdgcn_wmma_f32_16x16x32_bf16(false, qf[c].v, false, kf0.v,
                                                   (short)0, s0, false, false);
      s1 = __builtin_amdgcn_wmma_f32_16x16x32_bf16(false, qf[c].v, false, kf1.v,
                                                   (short)0, s1, false, false);
    }

    // online softmax in C-fragment layout (row = j+8h, cols across 16-lane half)
    float alpha[8];
#pragma unroll
    for (int j = 0; j < 8; ++j) {
      float mx = fmaxf(s0[j], s1[j]);
      mx = fmaxf(mx, __shfl_xor(mx, 1, 32));
      mx = fmaxf(mx, __shfl_xor(mx, 2, 32));
      mx = fmaxf(mx, __shfl_xor(mx, 4, 32));
      mx = fmaxf(mx, __shfl_xor(mx, 8, 32));
      float mn = fmaxf(mrow[j], mx);
      alpha[j] = __expf(mrow[j] - mn);
      mrow[j] = mn;
      float p0 = __expf(s0[j] - mn), p1 = __expf(s1[j] - mn);
      s0[j] = p0; s1[j] = p1;
      float sum = p0 + p1;
      sum += __shfl_xor(sum, 1, 32);
      sum += __shfl_xor(sum, 2, 32);
      sum += __shfl_xor(sum, 4, 32);
      sum += __shfl_xor(sum, 8, 32);
      lrow[j] = lrow[j] * alpha[j] + sum;
    }
#pragma unroll
    for (int t = 0; t < 8; ++t)
#pragma unroll
      for (int j = 0; j < 8; ++j) o[t][j] *= alpha[j];

    // bounce P through per-wave LDS slab to re-enter A-fragment layout
    unsigned short* pp = &sp[wave][0];
#pragma unroll
    for (int j = 0; j < 8; ++j) {
      pp[(j + 8 * h) * 32 + r]      = f32_bf16(s0[j]);
      pp[(j + 8 * h) * 32 + 16 + r] = f32_bf16(s1[j]);
    }
    Frag pf;
#pragma unroll
    for (int i = 0; i < 8; ++i) {
      int kk = 2 * i + ((i >= 4) ? 8 : 0) + 8 * h;
      pf.u[i] = *reinterpret_cast<const unsigned int*>(&pp[r * 32 + kk]);
    }
    // O += P @ V  (8 d-chunks of 16)
#pragma unroll
    for (int t = 0; t < 8; ++t) {
      Frag vf;
#pragma unroll
      for (int i = 0; i < 8; ++i) {
        unsigned int lo = sv[buf][(2 * i + ((i >= 4) ? 8 : 0) + 8 * h) * 128 + t * 16 + r];
        unsigned int hi = sv[buf][(2 * i + 1 + ((i >= 4) ? 8 : 0) + 8 * h) * 128 + t * 16 + r];
        vf.u[i] = lo | (hi << 16);
      }
      o[t] = __builtin_amdgcn_wmma_f32_16x16x32_bf16(false, pf.v, false, vf.v,
                                                     (short)0, o[t], false, false);
    }
    async_wait0();      // staged K/V tile (next buf) resident before handoff
    __syncthreads();
  }

#pragma unroll
  for (int t = 0; t < 8; ++t) {
    int d = t * 16 + r;
#pragma unroll
    for (int j = 0; j < 8; ++j) {
      int row = q0 + j + 8 * h;
      Out[(size_t)row * INNER + head * HDIM + d] = o[t][j] / lrow[j];
    }
  }
}

// ---------------------------------------------------------------- host
extern "C" void kernel_launch(void* const* d_in, const int* in_sizes, int n_in,
                              void* d_out, int out_size, void* d_ws, size_t ws_size,
                              hipStream_t stream) {
  (void)in_sizes; (void)n_in; (void)out_size; (void)ws_size;
  const float* hid  = (const float*)d_in[0];
  const float* enc  = (const float*)d_in[1];
  const float* ref  = (const float*)d_in[2];
  const float* rc   = (const float*)d_in[3];
  const float* rs   = (const float*)d_in[4];
  const float* crc  = (const float*)d_in[5];
  const float* crs  = (const float*)d_in[6];
  const float* Wq   = (const float*)d_in[7];  const float* bq   = (const float*)d_in[8];
  const float* Wk   = (const float*)d_in[9];  const float* bk   = (const float*)d_in[10];
  const float* Wv   = (const float*)d_in[11]; const float* bv   = (const float*)d_in[12];
  const float* Waq  = (const float*)d_in[13]; const float* baq  = (const float*)d_in[14];
  const float* Wak  = (const float*)d_in[15]; const float* bak  = (const float*)d_in[16];
  const float* Wav  = (const float*)d_in[17]; const float* bav  = (const float*)d_in[18];
  const float* Wout = (const float*)d_in[19]; const float* bout = (const float*)d_in[20];
  const float* Wadd = (const float*)d_in[21]; const float* badd = (const float*)d_in[22];
  const float* nq_w  = (const float*)d_in[23]; const float* nk_w  = (const float*)d_in[24];
  const float* naq_w = (const float*)d_in[25]; const float* nak_w = (const float*)d_in[26];
  const float* lkd = (const float*)d_in[27]; const float* lku = (const float*)d_in[28];
  const float* lvd = (const float*)d_in[29]; const float* lvu = (const float*)d_in[30];
  float* out = (float*)d_out;

  char* base = (char*)d_ws;
  size_t off = 0;
  auto alloc = [&](size_t bytes) -> void* {
    void* p = base + off;
    off = (off + bytes + 255) & ~(size_t)255;
    return p;
  };
  const size_t W33 = (size_t)INNER * INNER, W31 = (size_t)INNER * RANK;
  const size_t SH = (size_t)S_IMG * INNER, SE = (size_t)S_TXT * INNER;
  const size_t ST = (size_t)S_IMG * RANK;
  const size_t QSZ = (size_t)HEADS * SQ_TOT * HDIM, KSZ = (size_t)HEADS * SK_TOT * HDIM;
  const size_t AT = (size_t)SQ_TOT * INNER;

  unsigned short *Wq_b = (unsigned short*)alloc(W33 * 2), *Wk_b = (unsigned short*)alloc(W33 * 2);
  unsigned short *Wv_b = (unsigned short*)alloc(W33 * 2), *Waq_b = (unsigned short*)alloc(W33 * 2);
  unsigned short *Wak_b = (unsigned short*)alloc(W33 * 2), *Wav_b = (unsigned short*)alloc(W33 * 2);
  unsigned short *Wout_b = (unsigned short*)alloc(W33 * 2), *Wadd_b = (unsigned short*)alloc(W33 * 2);
  unsigned short *lkd_b = (unsigned short*)alloc(W31 * 2), *lku_b = (unsigned short*)alloc(W31 * 2);
  unsigned short *lvd_b = (unsigned short*)alloc(W31 * 2), *lvu_b = (unsigned short*)alloc(W31 * 2);
  unsigned short *hid_b = (unsigned short*)alloc(SH * 2), *enc_b = (unsigned short*)alloc(SE * 2);
  unsigned short *ref_b = (unsigned short*)alloc(SH * 2);
  unsigned short *t1b = (unsigned short*)alloc(ST * 2), *t2b = (unsigned short*)alloc(ST * 2);
  unsigned short *attn_b = (unsigned short*)alloc(AT * 2);
  unsigned short *Qb = (unsigned short*)alloc(QSZ * 2);
  unsigned short *KbP = (unsigned short*)alloc(KSZ * 2), *VbP = (unsigned short*)alloc(KSZ * 2);
  float *qf = (float*)alloc(SH * 4), *kf = (float*)alloc(SH * 4), *vf = (float*)alloc(SH * 4);
  float *eqf = (float*)alloc(SE * 4), *ekf = (float*)alloc(SE * 4), *evf = (float*)alloc(SE * 4);
  float *t1f = (float*)alloc(ST * 4), *t2f = (float*)alloc(ST * 4);
  float *rkf = (float*)alloc(SH * 4), *rvf = (float*)alloc(SH * 4);
  float *attnf = (float*)alloc(AT * 4);

  auto cvt = [&](const float* s, unsigned short* d, size_t n) {
    k_cvt_bf16<<<(unsigned)((n + 255) / 256), 256, 0, stream>>>(s, d, (long long)n);
  };
  auto gemm = [&](const unsigned short* A, const unsigned short* Bm, const float* bias,
                  float* C, int M, int N, int K) {
    dim3 g(N / 128, M / 128);
    k_gemm<<<g, 256, 0, stream>>>(A, Bm, bias, C, M, N, K);
  };
  auto pack = [&](const float* X, const float* w, const float* cs, const float* sn,
                  int ro, float eps, float scale, int S, int do_rms,
                  unsigned short* dst, int dseq, int doff) {
    int blocks = (S * HEADS + 7) / 8;
    k_pack<<<blocks, 256, 0, stream>>>(X, w, cs, sn, ro, eps, scale, S, do_rms, dst, dseq, doff);
  };

  // 1) f32 -> bf16
  cvt(Wq, Wq_b, W33);  cvt(Wk, Wk_b, W33);  cvt(Wv, Wv_b, W33);
  cvt(Waq, Waq_b, W33); cvt(Wak, Wak_b, W33); cvt(Wav, Wav_b, W33);
  cvt(Wout, Wout_b, W33); cvt(Wadd, Wadd_b, W33);
  cvt(lkd, lkd_b, W31); cvt(lku, lku_b, W31);
  cvt(lvd, lvd_b, W31); cvt(lvu, lvu_b, W31);
  cvt(hid, hid_b, SH);  cvt(enc, enc_b, SE);  cvt(ref, ref_b, SH);

  // 2) projections (bias fused, f32 out)
  gemm(hid_b, Wq_b, bq, qf, S_IMG, INNER, INNER);
  gemm(hid_b, Wk_b, bk, kf, S_IMG, INNER, INNER);
  gemm(hid_b, Wv_b, bv, vf, S_IMG, INNER, INNER);
  gemm(enc_b, Waq_b, baq, eqf, S_TXT, INNER, INNER);
  gemm(enc_b, Wak_b, bak, ekf, S_TXT, INNER, INNER);
  gemm(enc_b, Wav_b, bav, evf, S_TXT, INNER, INNER);
  // LoRA ref path
  gemm(ref_b, lkd_b, nullptr, t1f, S_REF, RANK, INNER);
  gemm(ref_b, lvd_b, nullptr, t2f, S_REF, RANK, INNER);
  cvt(t1f, t1b, ST);  cvt(t2f, t2b, ST);
  gemm(t1b, lku_b, nullptr, rkf, S_REF, INNER, RANK);
  gemm(t2b, lvu_b, nullptr, rvf, S_REF, INNER, RANK);

  // 3) RMS + RoPE + pack to [head][pos][128] bf16 (softmax scale folded into Q)
  const float SCL = 0.08838834764831845f; // 1/sqrt(128)
  pack(eqf, naq_w, rc, rs, 0,     1e-6f, SCL, S_TXT, 1, Qb, SQ_TOT, 0);
  pack(qf,  nq_w,  rc, rs, S_TXT, 1e-6f, SCL, S_IMG, 1, Qb, SQ_TOT, S_TXT);
  pack(ekf, nak_w, rc, rs, 0,     1e-6f, 1.f, S_TXT, 1, KbP, SK_TOT, 0);
  pack(kf,  nk_w,  rc, rs, S_TXT, 1e-6f, 1.f, S_IMG, 1, KbP, SK_TOT, S_TXT);
  pack(rkf, nullptr, crc, crs, 0, 1e-5f, 1.f, S_REF, 1, KbP, SK_TOT, SQ_TOT);
  pack(evf, nullptr, nullptr, nullptr, 0, 0.f, 1.f, S_TXT, 0, VbP, SK_TOT, 0);
  pack(vf,  nullptr, nullptr, nullptr, 0, 0.f, 1.f, S_IMG, 0, VbP, SK_TOT, S_TXT);
  pack(rvf, nullptr, nullptr, nullptr, 0, 0.f, 1.f, S_REF, 0, VbP, SK_TOT, SQ_TOT);

  // 4) flash attention -> attnf [pos][head*128+d]
  dim3 ag(SQ_TOT / 64, HEADS);
  k_attn<<<ag, 128, 0, stream>>>(Qb, KbP, VbP, attnf);

  // 5) output projections straight into d_out (hid_out first, then enc_out)
  cvt(attnf, attn_b, AT);
  gemm(attn_b + (size_t)S_TXT * INNER, Wout_b, bout, out, S_IMG, INNER, INNER);
  gemm(attn_b, Wadd_b, badd, out + (size_t)S_IMG * INNER, S_TXT, INNER, INNER);
}
